// SpatialBranch_drg_60842506715637
// MI455X (gfx1250) — matrix operand order repaired and checked by reading, not compiled, not run on hardware
//
#include <hip/hip_runtime.h>
#include <hip/hip_bf16.h>

typedef __attribute__((ext_vector_type(16))) _Float16 v16h;
typedef __attribute__((ext_vector_type(8)))  _Float16 v8h;
typedef __attribute__((ext_vector_type(8)))  float    v8f;

__device__ __forceinline__ v8f wmma16x16x32(v16h a, v16h b, v8f c) {
  return __builtin_amdgcn_wmma_f32_16x16x32_f16(false, a, false, b, (short)0, c, false, false);
}

// 16-half fragment from two aligned 16-byte LDS chunks (2x ds_load_b128).
__device__ __forceinline__ v16h ld_frag(const _Float16* p0, const _Float16* p1) {
  v16h r;
  ((v8h*)&r)[0] = *(const v8h*)p0;
  ((v8h*)&r)[1] = *(const v8h*)p1;
  return r;
}

// CDNA5 async global->LDS copy (ASYNCcnt-tracked, GVS addressing).
// lds_addr: wave-relative LDS byte address (generic LDS ptr truncated to 32b).
__device__ __forceinline__ void async_g2l_b64(unsigned lds_addr, int goff,
                                              const void* gbase) {
  asm volatile("global_load_async_to_lds_b64 %0, %1, %2"
               :: "v"(lds_addr), "v"(goff), "s"(gbase) : "memory");
}
__device__ __forceinline__ void wait_async0() {
  asm volatile("s_wait_asynccnt 0x0" ::: "memory");
}

#define NPAIR_PER_B 136
#define NPAIRS      1088        // 8 * 136

// ---------------------------------------------------------------------------
// Kernel 0: reorder conv2 weights into f16 K-slices w2h[ks][M][k_local]
// matching the GEMM K-order k = (ky*5+kx)*64 + ic, so the main kernel's
// per-K-step staging is a pure contiguous 2 KB async copy.
// ---------------------------------------------------------------------------
__global__ __launch_bounds__(256) void
sb_cvt_w2(const float* __restrict__ c2w, _Float16* __restrict__ w2h)
{
  const int idx = blockIdx.x * 256 + threadIdx.x;      // 50*32*32 = 51200
  if (idx >= 51200) return;
  const int ks = idx >> 10, rem = idx & 1023;
  const int M = rem >> 5, kl = rem & 31;
  w2h[idx] = (_Float16)c2w[M * 1600 + (kl + 32 * (ks & 1)) * 25 + (ks >> 1)];
}

// ---------------------------------------------------------------------------
// Kernel 1: per-pair fused pipeline, one workgroup (8 waves) per pair.
//   rasterize -> conv1(2->64)+pool (4-shift WMMA GEMM, B shared by 2 M-tiles)
//   -> conv2(64->32) WMMA GEMM: B fragments = 2x ds_load_b128 from h1T
//      [pixel][ch]; weight slices double-buffered via async global->LDS.
//   -> pool + mean -> feat32[pair][32].
// LDS 163072 B -> 2 pair-blocks per 320 KB WGP LDS.
// ---------------------------------------------------------------------------
__global__ __launch_bounds__(256) void
sb_fused_conv(const float* __restrict__ bboxes,   // (8,17,4)
              const float* __restrict__ c1w,      // (64,2,5,5)
              const float* __restrict__ c1b,      // (64)
              const _Float16* __restrict__ w2h,   // (50,32,32) reordered
              const float* __restrict__ c2b,      // (32)
              float* __restrict__ feat32)         // (1088,32) scratch
{
  // Region A time-multiplexed: [sp 16384 B | w1 8192 B] then [co 43264 B]
  __shared__ __align__(16) unsigned char smemA[43264];
  __shared__ __align__(16) _Float16 h1T[900 * 64];  // 115200 B [pixel][channel]
  __shared__ __align__(16) _Float16 w2s[2][1024];   // 4096 B double-buffered slices
  __shared__ float cb1[64];
  __shared__ float cb2[32];
  __shared__ float acc[32];

  _Float16* sp = (_Float16*)smemA;               // [2][64][64]
  _Float16* w1 = (_Float16*)(smemA + 16384);     // [64][64] (K zero-padded 50->64)
  _Float16* co = (_Float16*)smemA;               // [32][676] (phase 3)

  const int p = blockIdx.x;
  const int b = p / NPAIR_PER_B;
  int q = p % NPAIR_PER_B;
  int bi = 0;
  while (q >= 16 - bi) { q -= 16 - bi; ++bi; }
  const int bj = bi + 1 + q;

  const float* Bi = bboxes + (b * 17 + bi) * 4;
  const float* Bj = bboxes + (b * 17 + bj) * 4;
  const float bx0[2] = {Bi[0], Bj[0]}, by0[2] = {Bi[1], Bj[1]};
  const float bx1[2] = {Bi[2], Bj[2]}, by1[2] = {Bi[3], Bj[3]};

  const int t = threadIdx.x;
  const int lane = t & 31, wave = t >> 5;
  const int lm = lane & 15;      // A-row / B-col / D-col within a tile
  const int lh = lane >> 4;      // K-half selector

  // LDS byte addresses of the two weight-slice buffers (addr[31:0] = LDS offset)
  const unsigned w2s_lds0 = (unsigned)(unsigned long long)&w2s[0][0] + (unsigned)(t * 8);
  const unsigned w2s_lds1 = (unsigned)(unsigned long long)&w2s[1][0] + (unsigned)(t * 8);

  // kick off async copy of weight slice ks=0 (completes during conv1)
  async_g2l_b64(w2s_lds0, t * 8, (const void*)w2h);

  // ---- rasterize masks, stage conv1 weights + biases ----
  for (int idx = t; idx < 2 * 64 * 64; idx += 256) {
    const int ch = idx >> 12, y = (idx >> 6) & 63, x = idx & 63;
    const float fx = (float)x, fy = (float)y;
    const bool in = (fx >= bx0[ch]) && (fx < bx1[ch]) && (fy >= by0[ch]) && (fy < by1[ch]);
    sp[idx] = in ? (_Float16)1.0f : (_Float16)0.0f;
  }
  for (int idx = t; idx < 64 * 64; idx += 256) {
    const int oc = idx >> 6, k = idx & 63;
    w1[idx] = (k < 50) ? (_Float16)c1w[oc * 50 + k] : (_Float16)0.0f;
  }
  if (t < 64) cb1[t] = c1b[t];
  if (t < 32) { cb2[t] = c2b[t]; acc[t] = 0.0f; }
  __syncthreads();

  // =======================================================================
  // Stage 1: conv1 GEMM (M=64, N=900 pooled, K=64), pooling via 4 shifted
  // accumulators -> max. Each job = (M-tile pair, N-tile): one B gather
  // feeds two M-tiles. Unguarded loads: pad-K rows hit zero A-columns,
  // n>=900 columns discarded at the store.
  // =======================================================================
  int offT[2][16];
#pragma unroll
  for (int ks = 0; ks < 2; ++ks) {
#pragma unroll
    for (int i = 0; i < 16; ++i) {
      const int kk = ks * 32 + (i >> 3) * 16 + lh * 8 + (i & 7);
      int off = 0;
      if (kk < 50) {
        const int ic = kk / 25, r = kk % 25;
        off = ic * 4096 + (r / 5) * 64 + (r % 5);
      }
      offT[ks][i] = off;
    }
  }

  for (int job = wave; job < 2 * 57; job += 8) {
    const int mtp = job / 57;            // M-tile pair: rows mtp*32 .. mtp*32+31
    const int nt  = job % 57;
    const int n   = nt * 16 + lm;
    const int py  = n / 30, px = n % 30;
    const bool nok = (n < 900);
    const int sbase = nok ? ((2 * py) * 64 + 2 * px) : 0;

    v8f cacc[2][4] = {};
#pragma unroll
    for (int ks = 0; ks < 2; ++ks) {
      const int r0 = (mtp * 32 + lm) * 64 + ks * 32;
      const int r1 = (mtp * 32 + 16 + lm) * 64 + ks * 32;
      const v16h a0 = ld_frag(&w1[r0 + lh * 8], &w1[r0 + 16 + lh * 8]);
      const v16h a1 = ld_frag(&w1[r1 + lh * 8], &w1[r1 + 16 + lh * 8]);
#pragma unroll
      for (int s = 0; s < 4; ++s) {
        const int sb = sbase + (s >> 1) * 64 + (s & 1);
        v16h bm;
#pragma unroll
        for (int i = 0; i < 16; ++i) {
          bm[i] = sp[offT[ks][i] + sb];          // unconditional ds_load_u16
        }
        cacc[0][s] = wmma16x16x32(a0, bm, cacc[0][s]);
        cacc[1][s] = wmma16x16x32(a1, bm, cacc[1][s]);
      }
    }
    const int nn = nt * 16 + lm;
    if (nn < 900) {
#pragma unroll
      for (int u = 0; u < 2; ++u) {
        const int chb = mtp * 32 + u * 16 + 8 * lh;
        v8h hv;
#pragma unroll
        for (int r = 0; r < 8; ++r) {
          const float cm = fmaxf(fmaxf(cacc[u][0][r], cacc[u][1][r]),
                                 fmaxf(cacc[u][2][r], cacc[u][3][r]));
          hv[r] = (_Float16)(cm + cb1[chb + r]);
        }
        *(v8h*)&h1T[nn * 64 + chb] = hv;         // ds_store_b128
      }
    }
  }

  // =======================================================================
  // Stage 2: conv2 GEMM, K order k = (ky*5+kx)*64 + ic. B fragment = 16
  // contiguous channels of one pixel in h1T (2x ds_load_b128). Weight
  // slices arrive via double-buffered async global->LDS (one barrier/step).
  // =======================================================================
  const int mtw = wave & 1;
  int soffs[11];
#pragma unroll
  for (int ti = 0; ti < 11; ++ti) {
    const int tile = wave + ti * 8;
    const int n = (tile >> 1) * 16 + lm;
    soffs[ti] = (tile < 86 && n < 676) ? (n / 26) * 30 + (n % 26) : 0;
  }

  v8f cc[11] = {};
  for (int ks = 0; ks < 50; ++ks) {
    wait_async0();                   // my slice-ks load is in LDS
    __syncthreads();                 // everyone's is; ks-1 consumers done
    if (ks + 1 < 50) {               // prefetch next slice into other buffer
      async_g2l_b64((ks & 1) ? w2s_lds0 : w2s_lds1,
                    (ks + 1) * 2048 + t * 8, (const void*)w2h);
    }
    const _Float16* wbuf = &w2s[ks & 1][0];
    const v16h a = ld_frag(&wbuf[(mtw * 16 + lm) * 32 + lh * 16],
                           &wbuf[(mtw * 16 + lm) * 32 + lh * 16 + 8]);
    const int w  = ks >> 1;
    const int rowoff = (w / 5) * 30 + (w % 5);
    const int colb   = (ks & 1) * 32 + lh * 16;
#pragma unroll
    for (int ti = 0; ti < 11; ++ti) {
      const int tile = wave + ti * 8;
      if (tile < 86) {
        const int base = (soffs[ti] + rowoff) * 64 + colb;
        const v16h bf = ld_frag(&h1T[base], &h1T[base + 8]);
        cc[ti] = wmma16x16x32(a, bf, cc[ti]);
      }
    }
  }
  __syncthreads();                   // all WMMAs done before sp/w1 -> co reuse

  // spill C tiles (pre-pool conv2 output, f16)
#pragma unroll
  for (int ti = 0; ti < 11; ++ti) {
    const int tile = wave + ti * 8;
    if (tile < 86) {
      const int nt = tile >> 1;
#pragma unroll
      for (int r = 0; r < 8; ++r) {
        const int ch = mtw * 16 + r + 8 * lh;
        const int nn = nt * 16 + lm;
        if (nn < 676) co[ch * 676 + nn] = (_Float16)cc[ti][r];
      }
    }
  }
  __syncthreads();

  // 2x2 maxpool + spatial mean (+bias) -> acc[32]
  for (int cell = t; cell < 32 * 169; cell += 256) {
    const int ch = cell / 169, pi = cell % 169;
    const int py = pi / 13, px = pi % 13;
    const int b0 = ch * 676 + (2 * py) * 26 + 2 * px;
    const float m = fmaxf(fmaxf((float)co[b0], (float)co[b0 + 1]),
                          fmaxf((float)co[b0 + 26], (float)co[b0 + 27]));
    atomicAdd(&acc[ch], (m + cb2[ch]) * (1.0f / 169.0f));
  }
  __syncthreads();
  if (t < 32) feat32[p * 32 + t] = acc[t];
}

// ---------------------------------------------------------------------------
// Kernel 2: FC feat(1088x32) @ fc_w(512,32)^T + bias, ReLU. One tile/wave.
// ---------------------------------------------------------------------------
__global__ __launch_bounds__(256) void
sb_fc(const float* __restrict__ feat32, const float* __restrict__ fcw,
      const float* __restrict__ fcb, float* __restrict__ out)
{
  const int t = threadIdx.x, lane = t & 31, wave = t >> 5;
  const int job = blockIdx.x * 8 + wave;        // 2176 jobs exactly
  const int mt = job / 32, nt = job % 32;
  const int lm = lane & 15, lh = lane >> 4;

  const int M = mt * 16 + lm;
  v16h a, bm;
#pragma unroll
  for (int i = 0; i < 16; ++i) {
    const int k = lh * 16 + i;
    a[i] = (_Float16)feat32[M * 32 + k];
  }
  const int n = nt * 16 + lm;
#pragma unroll
  for (int i = 0; i < 16; ++i) {
    bm[i] = (_Float16)fcw[n * 32 + lh * 16 + i];
  }
  v8f c = {};
  c = wmma16x16x32(a, bm, c);
#pragma unroll
  for (int r = 0; r < 8; ++r) {
    const int m  = mt * 16 + r + 8 * lh;
    const int nn = nt * 16 + lm;
    const float v = c[r] + fcb[nn];
    out[m * 512 + nn] = v > 0.0f ? v : 0.0f;
  }
}

// ---------------------------------------------------------------------------
// Kernel 3: slicing triples (b, i, j) appended after feat in d_out.
// ---------------------------------------------------------------------------
__global__ void sb_slicing(float* __restrict__ out)
{
  const int p = blockIdx.x * blockDim.x + threadIdx.x;
  if (p >= NPAIRS) return;
  const int b = p / NPAIR_PER_B;
  int q = p % NPAIR_PER_B;
  int bi = 0;
  while (q >= 16 - bi) { q -= 16 - bi; ++bi; }
  const int bj = bi + 1 + q;
  out[p * 3 + 0] = (float)b;
  out[p * 3 + 1] = (float)bi;
  out[p * 3 + 2] = (float)bj;
}

extern "C" void kernel_launch(void* const* d_in, const int* in_sizes, int n_in,
                              void* d_out, int out_size, void* d_ws, size_t ws_size,
                              hipStream_t stream) {
  const float* bboxes = (const float*)d_in[0];
  const float* c1w    = (const float*)d_in[1];
  const float* c1b    = (const float*)d_in[2];
  const float* c2w    = (const float*)d_in[3];
  const float* c2b    = (const float*)d_in[4];
  const float* fcw    = (const float*)d_in[5];
  const float* fcb    = (const float*)d_in[6];

  float*     out    = (float*)d_out;
  float*     feat32 = (float*)d_ws;                          // 139264 B
  _Float16*  w2h    = (_Float16*)((char*)d_ws + 1088 * 32 * 4); // 102400 B

  sb_cvt_w2<<<200, 256, 0, stream>>>(c2w, w2h);
  sb_fused_conv<<<NPAIRS, 256, 0, stream>>>(bboxes, c1w, c1b, w2h, c2b, feat32);
  sb_fc<<<272, 256, 0, stream>>>(feat32, fcw, fcb, out);
  sb_slicing<<<(NPAIRS + 255) / 256, 256, 0, stream>>>(out + NPAIRS * 512);
}
